// RealBounce_46239617909491
// MI455X (gfx1250) — matrix operand order repaired and checked by reading, not compiled, not run on hardware
//
#include <hip/hip_runtime.h>
#include <hip/hip_bf16.h>

typedef __attribute__((ext_vector_type(16))) _Float16 v16h;
typedef __attribute__((ext_vector_type(8)))  _Float16 v8h;
typedef __attribute__((ext_vector_type(8)))  float    v8f;

#define NPTS        65536
#define MRAYS       16
#define APPF        48
#define HIDW        128
#define PTS_PER_BLK 16
#define THREADS     128

// ---- LDS layout (bytes, all 32B aligned) ----
#define OFF_W1    0       // swizzled B-operands: 16 ops x 32 lanes x 32B = 16384 (Wd1 then Wb1)
#define OFF_W2    16384   // swizzled B-operands: 4 ops x 32 lanes x 32B  = 4096  (Wd2 then Wb2)
#define OFF_B1    20480   // 128 f32 = 512
#define OFF_B2    20992   // 16  f32 = 64
#define OFF_BIN   21056   // 128x64 f16 = 16384 (also overlays 16x64 diffuse A tile)
#define OFF_HID   37440   // 4 waves x 16x128 f16 = 16384
#define OFF_INC   53824   // 128x3 f32 = 1536 (incoming, pre-masked)
#define OFF_DOUT  55360   // 16x16 f32 = 1024 (diffuse head output)
#define OFF_CNT   56384   // 8 f32 = 32
#define OFF_NF    56416   // 16x48 f16 = 1536 (per-point noisy features)
#define SMEM_BYTES 57952

__device__ __forceinline__ float sigm(float x) { return 1.0f / (1.0f + expf(-x)); }

// A operand: 16x32 f16 tile from row-major LDS (stride `strideH` halves).
// Contiguous K order (matches swizzled B): M=lane&15, halves 0..15 -> K 16*hi..16*hi+15.
__device__ __forceinline__ v16h load_a_f16(const _Float16* base, int strideH, int kOff, int lane) {
  const int m = lane & 15, hi = lane >> 4;
  const _Float16* p = base + m * strideH + kOff + 16 * hi;
  v8h lo = *(const v8h*)p;          // 16B aligned
  v8h hh = *(const v8h*)(p + 8);    // +16B
  return __builtin_shufflevector(lo, hh, 0, 1, 2, 3, 4, 5, 6, 7, 8, 9, 10, 11, 12, 13, 14, 15);
}

// B operand from pre-swizzled LDS: operand `op`, 32B contiguous per lane.
__device__ __forceinline__ v16h load_b_sw(const _Float16* sw, int op, int lane) {
  return ((const v16h*)sw)[op * 32 + lane];
}

__device__ __forceinline__ v8f wmma_f16(v16h a, v16h b, v8f c) {
  return __builtin_amdgcn_wmma_f32_16x16x32_f16(false, a, false, b, (short)0, c, false, false);
}

__global__ __launch_bounds__(THREADS)
void realbounce_fused(const float* __restrict__ xyzs,
                      const float* __restrict__ app_features,
                      const float* __restrict__ viewdirs,
                      const float* __restrict__ normals,
                      const float* __restrict__ feat_noise,
                      const float* __restrict__ uu,
                      const float* __restrict__ Wd1, const float* __restrict__ bd1,
                      const float* __restrict__ Wd2, const float* __restrict__ bd2,
                      const float* __restrict__ Wb1, const float* __restrict__ bb1,
                      const float* __restrict__ Wb2, const float* __restrict__ bb2,
                      const unsigned char* __restrict__ ray_mask,
                      float* __restrict__ out) {
  __shared__ alignas(32) unsigned char smem[SMEM_BYTES];
  _Float16* sW1  = (_Float16*)(smem + OFF_W1);
  _Float16* sW2  = (_Float16*)(smem + OFF_W2);
  float*    sB1  = (float*)(smem + OFF_B1);
  float*    sB2  = (float*)(smem + OFF_B2);
  _Float16* sBin = (_Float16*)(smem + OFF_BIN);
  _Float16* sHid = (_Float16*)(smem + OFF_HID);
  float*    sInc = (float*)(smem + OFF_INC);
  float*    sDout= (float*)(smem + OFF_DOUT);
  float*    sCnt = (float*)(smem + OFF_CNT);
  _Float16* sNF  = (_Float16*)(smem + OFF_NF);

  const int tid  = threadIdx.x;
  const int lane = tid & 31;
  const int wave = tid >> 5;
  const int blk  = blockIdx.x;

  // ---------- Stage 1: diffuse weights -> swizzled B-operand layout; build A tile ----
  // W1 swizzle: operand op = j*2+ks; dword i of lane ln holds (W[k0][n], W[k0+1][n]),
  // k0 = 32*ks + 16*(ln>>4) + 2*i, n = 16*j + (ln&15).
  for (int idx = tid; idx < 16 * 256; idx += THREADS) {   // 4096 dwords
    const int op = idx >> 8, ln = (idx >> 3) & 31, i = idx & 7;
    const int j = op >> 1, ks = op & 1;
    const int k0 = 32 * ks + 16 * (ln >> 4) + 2 * i;
    const int n  = 16 * j + (ln & 15);
    union { unsigned int u; _Float16 h[2]; } t;
    t.h[0] = (_Float16)((k0     < 54) ? Wd1[(k0    ) * 128 + n] : 0.0f);
    t.h[1] = (_Float16)((k0 + 1 < 54) ? Wd1[(k0 + 1) * 128 + n] : 0.0f);
    ((unsigned int*)sW1)[idx] = t.u;
  }
  for (int idx = tid; idx < 4 * 256; idx += THREADS) {    // 1024 dwords
    const int op = idx >> 8, ln = (idx >> 3) & 31, i = idx & 7;
    const int k0 = 32 * op + 16 * (ln >> 4) + 2 * i;
    const int n  = ln & 15;
    union { unsigned int u; _Float16 h[2]; } t;
    t.h[0] = (_Float16)((n < 9) ? Wd2[(k0    ) * 9 + n] : 0.0f);
    t.h[1] = (_Float16)((n < 9) ? Wd2[(k0 + 1) * 9 + n] : 0.0f);
    ((unsigned int*)sW2)[idx] = t.u;
  }
  if (tid < 128) sB1[tid] = bd1[tid];
  if (tid < 16)  sB2[tid] = (tid < 9) ? bd2[tid] : 0.0f;
  for (int idx = tid; idx < 16 * 64; idx += THREADS) {    // diffuse A tile overlays sBin
    const int pr = idx >> 6, k = idx & 63;
    const int gp = blk * PTS_PER_BLK + pr;
    float v = 0.0f;
    if (k < 3)       v = xyzs[gp * 3 + k];
    else if (k < 6)  v = viewdirs[gp * 3 + k - 3];
    else if (k < 54) v = app_features[gp * APPF + k - 6];
    sBin[idx] = (_Float16)v;
  }
  __syncthreads();

  // ---------- Stage 2: diffuse MLP (wave 0) ------------------------------------------
  if (wave == 0) {
    _Float16* sH = sHid;
    const int nlo = lane & 15, hi = lane >> 4;
    const v16h A0 = load_a_f16(sBin, 64, 0, lane);
    const v16h A1 = load_a_f16(sBin, 64, 32, lane);
#pragma unroll
    for (int j = 0; j < 8; ++j) {
      v8f acc = {0, 0, 0, 0, 0, 0, 0, 0};
      acc = wmma_f16(A0, load_b_sw(sW1, j * 2 + 0, lane), acc);
      acc = wmma_f16(A1, load_b_sw(sW1, j * 2 + 1, lane), acc);
      const int nh = j * 16 + nlo;
      const float bb = sB1[nh];
#pragma unroll
      for (int r = 0; r < 8; ++r)
        sH[(r + 8 * hi) * 128 + nh] = (_Float16)fmaxf(acc[r] + bb, 0.0f);
    }
    v8f o = {0, 0, 0, 0, 0, 0, 0, 0};
#pragma unroll
    for (int ks = 0; ks < 4; ++ks)
      o = wmma_f16(load_a_f16(sH, 128, 32 * ks, lane), load_b_sw(sW2, ks, lane), o);
    const float b2v = sB2[nlo];
#pragma unroll
    for (int r = 0; r < 8; ++r)
      sDout[(r + 8 * hi) * 16 + nlo] = o[r] + b2v;
  }
  __syncthreads();

  // ---------- Stage 3: overlay bounce weights (swizzled) + per-point noisy feats -----
  for (int idx = tid; idx < 16 * 256; idx += THREADS) {
    const int op = idx >> 8, ln = (idx >> 3) & 31, i = idx & 7;
    const int j = op >> 1, ks = op & 1;
    const int k0 = 32 * ks + 16 * (ln >> 4) + 2 * i;
    const int n  = 16 * j + (ln & 15);
    union { unsigned int u; _Float16 h[2]; } t;
    t.h[0] = (_Float16)Wb1[(k0    ) * 128 + n];
    t.h[1] = (_Float16)Wb1[(k0 + 1) * 128 + n];
    ((unsigned int*)sW1)[idx] = t.u;
  }
  for (int idx = tid; idx < 4 * 256; idx += THREADS) {
    const int op = idx >> 8, ln = (idx >> 3) & 31, i = idx & 7;
    const int k0 = 32 * op + 16 * (ln >> 4) + 2 * i;
    const int n  = ln & 15;
    union { unsigned int u; _Float16 h[2]; } t;
    t.h[0] = (_Float16)((n < 3) ? Wb2[(k0    ) * 3 + n] : 0.0f);
    t.h[1] = (_Float16)((n < 3) ? Wb2[(k0 + 1) * 3 + n] : 0.0f);
    ((unsigned int*)sW2)[idx] = t.u;
  }
  if (tid < 128) sB1[tid] = bb1[tid];
  if (tid < 16)  sB2[tid] = (tid < 3) ? bb2[tid] : 0.0f;
  for (int idx = tid; idx < 16 * APPF; idx += THREADS) {
    const int p = idx / APPF, j = idx - p * APPF;
    const int gp = blk * PTS_PER_BLK + p;
    sNF[idx] = (_Float16)(app_features[gp * APPF + j] + 0.01f * feat_noise[gp * APPF + j]);
  }
  __syncthreads();

  // Preload all bounce B operands into registers (per wave): 16 + 4 v16h.
  v16h Bw1[16], Bw2[4];
#pragma unroll
  for (int op = 0; op < 16; ++op) Bw1[op] = load_b_sw(sW1, op, lane);
#pragma unroll
  for (int op = 0; op < 4; ++op)  Bw2[op] = load_b_sw(sW2, op, lane);

  // ---------- Stage 4: two half-passes of 8 points x 16 rays --------------------------
  for (int hp = 0; hp < 2; ++hp) {
    // ---- per-ray geometry: one thread = one ray ----
    {
      const int p  = tid >> 4;
      const int m  = tid & 15;
      const int pf = hp * 8 + p;
      const int gp = blk * PTS_PER_BLK + pf;

      float nx = normals[gp * 3 + 0], ny = normals[gp * 3 + 1], nz = normals[gp * 3 + 2];
      const float nl = sqrtf(nx * nx + ny * ny + nz * nz) + 1e-8f;
      const float bNx = nx / nl, bNy = ny / nl, bNz = nz / nl;
      const float bVx = -viewdirs[gp * 3 + 0], bVy = -viewdirs[gp * 3 + 1], bVz = -viewdirs[gp * 3 + 2];

      float upx, upy, upz;
      if (fabsf(bNz) < 0.99f) { upx = 0.0f; upy = 0.0f; upz = 1.0f; }
      else                    { upx = 1.0f; upy = 0.0f; upz = 0.0f; }
      float tx = upy * bNz - upz * bNy;
      float ty = upz * bNx - upx * bNz;
      float tz = upx * bNy - upy * bNx;
      const float tl = sqrtf(tx * tx + ty * ty + tz * tz) + 1e-8f;
      tx /= tl; ty /= tl; tz /= tl;
      const float bx = bNy * tz - bNz * ty;
      const float by = bNz * tx - bNx * tz;
      const float bz = bNx * ty - bNy * tx;

      const float r1 = sigm(sDout[pf * 16 + 6]) * 0.98f + 0.01f;
      float a2 = r1 * r1; a2 = a2 * a2;
      const float a2c = fmaxf(a2, 1e-6f);
      const float expfac = expf(0.1f * logf(a2c));

      const float u1 = uu[(gp * MRAYS + m) * 2 + 0];
      const float u2 = uu[(gp * MRAYS + m) * 2 + 1];
      const float ct = sqrtf((1.0f - u1) / (1.0f + (a2 - 1.0f) * u1));
      const float st = sqrtf(fmaxf(1.0f - ct * ct, 0.0f));
      const float phi = 6.28318530717958647692f * u2;
      const float hl0 = st * cosf(phi), hl1 = st * sinf(phi), hl2 = ct;

      const float Hx = tx * hl0 + bx * hl1 + bNx * hl2;
      const float Hy = ty * hl0 + by * hl1 + bNy * hl2;
      const float Hz = tz * hl0 + bz * hl1 + bNz * hl2;
      const float vdh = bVx * Hx + bVy * Hy + bVz * Hz;
      float Lx = 2.0f * vdh * Hx - bVx;
      float Ly = 2.0f * vdh * Hy - bVy;
      float Lz = 2.0f * vdh * Hz - bVz;
      const float Ll = sqrtf(Lx * Lx + Ly * Ly + Lz * Lz) + 1e-8f;
      Lx /= Ll; Ly /= Ll; Lz /= Ll;

      const float h0 = tx * Hx + ty * Hy + tz * Hz;
      const float h1 = bx * Hx + by * Hy + bz * Hz;
      const float h2 = bNx * Hx + bNy * Hy + bNz * Hz;
      const float d0 = tx * Lx + ty * Ly + tz * Lz;
      const float d1 = bx * Lx + by * Ly + bz * Lz;
      const float d2 = bNx * Lx + bNy * Ly + bNz * Lz;

      const float maskf = ray_mask[gp * MRAYS + m] ? 1.0f : 0.0f;
      const float lz = fmaxf(Lz, 0.0f);
      sInc[tid * 3 + 0] = (lz * 1.0f + 0.1f) * expfac * maskf;
      sInc[tid * 3 + 1] = (lz * 0.9f + 0.1f) * expfac * maskf;
      sInc[tid * 3 + 2] = (lz * 0.8f + 0.1f) * expfac * maskf;

      // Build 64-half bin row in registers, then 8x ds_store_b128.
      _Float16 h64[64];
      h64[0] = (_Float16)bVx;  h64[1] = (_Float16)bVy;  h64[2] = (_Float16)bVz;
      h64[3] = (_Float16)Lx;   h64[4] = (_Float16)Ly;   h64[5] = (_Float16)Lz;
      h64[6] = (_Float16)bNx;  h64[7] = (_Float16)bNy;  h64[8] = (_Float16)bNz;
      h64[9] = (_Float16)h0;   h64[10] = (_Float16)h1;  h64[11] = (_Float16)h2;
      h64[12] = (_Float16)d0;  h64[13] = (_Float16)d1;  h64[14] = (_Float16)d2;
      const v8h* nfp = (const v8h*)(sNF + pf * APPF);
#pragma unroll
      for (int c = 0; c < 6; ++c) {
        v8h v = nfp[c];
#pragma unroll
        for (int e = 0; e < 8; ++e) h64[15 + c * 8 + e] = v[e];
      }
      h64[63] = (_Float16)r1;
      v8h* dst = (v8h*)(sBin + tid * 64);
#pragma unroll
      for (int c = 0; c < 8; ++c) {
        v8h chunk;
#pragma unroll
        for (int e = 0; e < 8; ++e) chunk[e] = h64[c * 8 + e];
        dst[c] = chunk;
      }

      if (m == 0) {
        float c = 0.0f;
        for (int mm = 0; mm < MRAYS; ++mm)
          c += ray_mask[gp * MRAYS + mm] ? 1.0f : 0.0f;
        sCnt[p] = c;
      }
    }
    __syncthreads();

    // ---- bounce MLP: each wave handles 2 row-tiles (1 point = 16 rays each) ----
    for (int t = wave; t < 8; t += 4) {
      const _Float16* rows = sBin + t * 16 * 64;
      _Float16* sH = sHid + wave * (16 * 128);
      const int nlo = lane & 15, hi = lane >> 4;
      const v16h A0 = load_a_f16(rows, 64, 0, lane);
      const v16h A1 = load_a_f16(rows, 64, 32, lane);
#pragma unroll
      for (int j = 0; j < 8; ++j) {
        v8f acc = {0, 0, 0, 0, 0, 0, 0, 0};
        acc = wmma_f16(A0, Bw1[j * 2 + 0], acc);
        acc = wmma_f16(A1, Bw1[j * 2 + 1], acc);
        const int nh = j * 16 + nlo;
        const float bb = sB1[nh];
#pragma unroll
        for (int r = 0; r < 8; ++r)
          sH[(r + 8 * hi) * 128 + nh] = (_Float16)fmaxf(acc[r] + bb, 0.0f);
      }
      v8f o = {0, 0, 0, 0, 0, 0, 0, 0};
#pragma unroll
      for (int ks = 0; ks < 4; ++ks)
        o = wmma_f16(load_a_f16(sH, 128, 32 * ks, lane), Bw2[ks], o);

      const float b2v = sB2[nlo];
      float partial = 0.0f;
#pragma unroll
      for (int r = 0; r < 8; ++r) {
        const int ray = r + 8 * hi;
        const float bw = sigm(o[r] + b2v);
        const float inc = (nlo < 3) ? sInc[(t * 16 + ray) * 3 + nlo] : 0.0f;
        partial += bw * inc;
      }
      partial += __shfl_xor(partial, 16, 32);
      if (hi == 0 && nlo < 3) {
        const int pf = hp * 8 + t;
        const float refl = partial / (sCnt[t] + 1e-8f);
        const float dif  = sigm(sDout[pf * 16 + nlo]);
        out[(blk * PTS_PER_BLK + pf) * 3 + nlo] = refl + dif;
      }
    }
    __syncthreads();
  }
}

extern "C" void kernel_launch(void* const* d_in, const int* in_sizes, int n_in,
                              void* d_out, int out_size, void* d_ws, size_t ws_size,
                              hipStream_t stream) {
  const float* xyzs         = (const float*)d_in[0];
  const float* app_features = (const float*)d_in[1];
  const float* viewdirs     = (const float*)d_in[2];
  const float* normals      = (const float*)d_in[3];
  const float* feat_noise   = (const float*)d_in[4];
  const float* uu           = (const float*)d_in[5];
  const float* Wd1          = (const float*)d_in[6];
  const float* bd1          = (const float*)d_in[7];
  const float* Wd2          = (const float*)d_in[8];
  const float* bd2          = (const float*)d_in[9];
  const float* Wb1          = (const float*)d_in[10];
  const float* bb1          = (const float*)d_in[11];
  const float* Wb2          = (const float*)d_in[12];
  const float* bb2          = (const float*)d_in[13];
  // d_in[14] = bounce_mask (already folded into ray_mask by setup)
  const unsigned char* ray_mask = (const unsigned char*)d_in[15];
  float* out = (float*)d_out;

  dim3 grid(NPTS / PTS_PER_BLK);
  dim3 block(THREADS);
  realbounce_fused<<<grid, block, 0, stream>>>(xyzs, app_features, viewdirs, normals,
                                               feat_noise, uu, Wd1, bd1, Wd2, bd2,
                                               Wb1, bb1, Wb2, bb2, ray_mask, out);
}
